// MPNN_62689342653101
// MI455X (gfx1250) — matrix operand order repaired and checked by reading, not compiled, not run on hardware
//
#include <hip/hip_runtime.h>
#include <hip/hip_bf16.h>

// ---------------------------------------------------------------------------
// MPNN (NNConv + GRU) for MI455X / gfx1250, wave32.
// Heavy math: msg[e,o] = sum_{k,i} h1[e,k]*xsrc[e,i]*W2[k,i*64+o]
//   => GEMM  P[E,4160] @ W2''[4160,64]  (K = 4096 + 64 bias rows),
//      P built on the fly (outer product) per 32-edge tile,
//      v_wmma_f32_16x16x32_bf16 with register-double-buffered B fragments
//      and v_pk_mul_bf16 A-fragment construction.
// ---------------------------------------------------------------------------

typedef __attribute__((ext_vector_type(16))) __bf16 v16bf;
typedef __attribute__((ext_vector_type(8)))  float  v8f;

#define D64 64
#define FNN 92
#define FEE 50
#define NL  3
#define KB_TOT 130   // 128 k-blocks of W2 + 2 blocks for b2 bias rows
#define PB_PER_LAYER (KB_TOT * 4 * 32 * 8)  // dwords: kb * nt * lane * vgpr

__device__ __forceinline__ float bf2f(unsigned short h) {
    unsigned u = ((unsigned)h) << 16;
    float f; __builtin_memcpy(&f, &u, 4); return f;
}
__device__ __forceinline__ unsigned short f2bf(float f) {
    unsigned u; __builtin_memcpy(&u, &f, 4);
    unsigned r = u + 0x7fffu + ((u >> 16) & 1u);   // RNE
    return (unsigned short)(r >> 16);
}
// Packed bf16 * bf16 multiply: one VOP3P instruction (CDNA5 V_PK_MUL_BF16).
__device__ __forceinline__ unsigned pkmul_bf16(unsigned x, unsigned s) {
    unsigned d;
    asm("v_pk_mul_bf16 %0, %1, %2" : "=v"(d) : "v"(x), "v"(s));
    return d;
}

union AFrag { v16bf v; unsigned u[8]; uint4 q[2]; };

// ---------------------------------------------------------------------------
__global__ void zero_kernel(float* p, size_t n) {
    size_t i = (size_t)blockIdx.x * blockDim.x + threadIdx.x;
    if (i < n) p[i] = 0.0f;
}

// pre-FC: h = relu(x @ pre_w + pre_b), x[N,92], pre_w[92,64]
__global__ void __launch_bounds__(64)
prefc_kernel(const float* __restrict__ x, const float* __restrict__ w,
             const float* __restrict__ b, float* __restrict__ h, int N) {
    __shared__ float sx[FNN];
    int n = blockIdx.x; if (n >= N) return;
    int o = threadIdx.x;
    for (int f = o; f < FNN; f += 64) sx[f] = x[(size_t)n * FNN + f];
    __syncthreads();
    float acc = b[o];
    for (int f = 0; f < FNN; ++f) acc += sx[f] * w[f * 64 + o];
    h[(size_t)n * 64 + o] = fmaxf(acc, 0.0f);
}

// Pack W2''[4160,64] (fp32 edge_w2 + edge_b2) into bf16 WMMA B fragments.
// B 32x16 fragment k order used here (A uses the same): VGPR v, lane(n=l&15, kh=l>>4): K = kh*16+2v (+1).
__global__ void pack_w2_kernel(const float* __restrict__ ew2,
                               const float* __restrict__ eb2,
                               unsigned* __restrict__ PB) {
    int t = blockIdx.x * blockDim.x + threadIdx.x;
    if (t >= NL * PB_PER_LAYER) return;
    int l = t / PB_PER_LAYER, r = t % PB_PER_LAYER;
    int kb = r >> 10;  int r2 = r & 1023;
    int nt = r2 >> 8;  int r3 = r2 & 255;
    int lane = r3 >> 3; int v = r3 & 7;
    int kh = lane >> 4; int n = nt * 16 + (lane & 15);
    int g = kb * 32 + kh * 16 + 2 * v;
    float v0, v1;
    {
        int gg = g;
        if (gg < 4096) { int k = gg >> 6, i = gg & 63; v0 = ew2[((size_t)(l * 64 + k)) * 4096 + i * 64 + n]; }
        else           { int i = gg - 4096;           v0 = eb2[(size_t)l * 4096 + i * 64 + n]; }
        gg = g + 1;
        if (gg < 4096) { int k = gg >> 6, i = gg & 63; v1 = ew2[((size_t)(l * 64 + k)) * 4096 + i * 64 + n]; }
        else           { int i = gg - 4096;           v1 = eb2[(size_t)l * 4096 + i * 64 + n]; }
    }
    PB[t] = (unsigned)f2bf(v0) | ((unsigned)f2bf(v1) << 16);
}

// Edge MLP layer 1 + source-node gather, both emitted as bf16 rows.
__global__ void __launch_bounds__(64)
edge_pre_kernel(const float* __restrict__ ea, const float* __restrict__ w1,
                const float* __restrict__ b1, const int* __restrict__ src,
                const float* __restrict__ h, unsigned short* __restrict__ h1b,
                unsigned short* __restrict__ xsb, int E) {
    __shared__ float se[FEE];
    int e = blockIdx.x; if (e >= E) return;
    int o = threadIdx.x;
    if (o < FEE) se[o] = ea[(size_t)e * FEE + o];
    __syncthreads();
    float acc = b1[o];
    for (int f = 0; f < FEE; ++f) acc += se[f] * w1[f * 64 + o];
    h1b[(size_t)e * 64 + o] = f2bf(fmaxf(acc, 0.0f));
    int s = src[e];
    xsb[(size_t)e * 64 + o] = f2bf(h[(size_t)s * 64 + o]);
}

__global__ void deg_kernel(const int* __restrict__ dst, int* __restrict__ deg, int E) {
    int e = blockIdx.x * blockDim.x + threadIdx.x;
    if (e < E) atomicAdd(&deg[dst[e]], 1);
}

// ---- The WMMA GEMM: one wave per 32-edge tile (2 M-tiles), K = 130x32. ----
__global__ void __launch_bounds__(32)
msg_wmma_kernel(const unsigned short* __restrict__ h1b,
                const unsigned short* __restrict__ xsb,
                const unsigned* __restrict__ PB,
                const int* __restrict__ dst,
                float* __restrict__ agg, int E) {
    __shared__ uint4 sXS[256];   // 32 rows x 64 bf16 = 4KB
    __shared__ uint4 sH1[256];
    int e0 = blockIdx.x * 32;
    if (e0 >= E) return;
    const int lane = threadIdx.x;
    const int m = lane & 15, half = lane >> 4;

    // stage 32 rows of xs / h1 (row-clamped for ragged tail)
    {
        int er = e0 + lane; if (er >= E) er = E - 1;
        const uint4* gxr = (const uint4*)(xsb + (size_t)er * 64);
        const uint4* ghr = (const uint4*)(h1b + (size_t)er * 64);
#pragma unroll
        for (int j = 0; j < 8; ++j) { sXS[lane * 8 + j] = gxr[j]; sH1[lane * 8 + j] = ghr[j]; }
    }
    __syncthreads();
    const unsigned short* h1s = (const unsigned short*)sH1;

    v8f acc[2][4];
#pragma unroll
    for (int t = 0; t < 2; ++t)
#pragma unroll
        for (int nt = 0; nt < 4; ++nt)
#pragma unroll
            for (int i = 0; i < 8; ++i) acc[t][nt][i] = 0.0f;

    AFrag bcur[4], bnext[4];

    auto loadB = [&](AFrag* b, int kb) {
#pragma unroll
        for (int nt = 0; nt < 4; ++nt) {
            const uint4* p = (const uint4*)PB + (((size_t)(kb * 4 + nt)) * 32 + lane) * 2;
            b[nt].q[0] = p[0]; b[nt].q[1] = p[1];
        }
    };
    // One K-step: build 2 A fragments (outer-product rows), 8 WMMAs.
    auto step = [&](int kb, AFrag* b, unsigned hh0, unsigned hh1) {
        const int i0q = (kb & 1) << 2;              // uint4 offset: (i0>>3)
#pragma unroll
        for (int t = 0; t < 2; ++t) {
            const int xi = (m + 16 * t) * 8 + i0q + 2 * half;
            uint4 r1 = sXS[xi];
            uint4 r2 = sXS[xi + 1];
            unsigned hh = t ? hh1 : hh0;
            AFrag a;
            a.u[0] = pkmul_bf16(r1.x, hh); a.u[1] = pkmul_bf16(r1.y, hh);
            a.u[2] = pkmul_bf16(r1.z, hh); a.u[3] = pkmul_bf16(r1.w, hh);
            a.u[4] = pkmul_bf16(r2.x, hh); a.u[5] = pkmul_bf16(r2.y, hh);
            a.u[6] = pkmul_bf16(r2.z, hh); a.u[7] = pkmul_bf16(r2.w, hh);
#pragma unroll
            for (int nt = 0; nt < 4; ++nt)
                acc[t][nt] = __builtin_amdgcn_wmma_f32_16x16x32_bf16(
                    false, a.v, false, b[nt].v, (short)0, acc[t][nt], false, false);
        }
    };
    auto geth = [&](int row, int k0) -> unsigned {
        unsigned hu = h1s[row * 64 + k0];
        return hu | (hu << 16);
    };

    loadB(bcur, 0);
    for (int kb = 0; kb < 128; kb += 2) {
        const int k0 = kb >> 1;
        unsigned hh0 = geth(m, k0), hh1 = geth(m + 16, k0);
        loadB(bnext, kb + 1);                 // overlaps WMMAs on bcur
        step(kb, bcur, hh0, hh1);
        loadB(bcur, kb + 2);                  // kb+2 <= 128 < KB_TOT
        step(kb + 1, bnext, hh0, hh1);
        if (kb + 4 < KB_TOT)
            __builtin_prefetch((const uint4*)PB + (((size_t)((kb + 4) * 4)) * 32 + lane) * 2, 0, 3);
    }
    // bias rows (implicit h1 = 1.0): kb = 128, 129
    loadB(bnext, 129);
    step(128, bcur, 0x3F803F80u, 0x3F803F80u);
    step(129, bnext, 0x3F803F80u, 0x3F803F80u);

    // C/D layout: VGPR v -> row M = v + half*8 (+16*t), lane&15 -> column in nt tile
    const int n = lane & 15;
#pragma unroll
    for (int t = 0; t < 2; ++t)
#pragma unroll
        for (int v = 0; v < 8; ++v) {
            int e = e0 + t * 16 + half * 8 + v;
            if (e < E) {
                float* row = agg + (size_t)dst[e] * 64;
#pragma unroll
                for (int nt = 0; nt < 4; ++nt)
                    unsafeAtomicAdd(row + nt * 16 + n, acc[t][nt][v]);
            }
        }
}

// m = agg/deg + h@root + conv_bias ; accumulate BN batch stats.
__global__ void __launch_bounds__(64)
node_m_kernel(const float* __restrict__ agg, const int* __restrict__ deg,
              const float* __restrict__ h, const float* __restrict__ root,
              const float* __restrict__ cbias, float* __restrict__ mbuf,
              float* __restrict__ bnsum, float* __restrict__ bnsq, int N) {
    __shared__ float sh[64];
    int o = threadIdx.x;
    float ls = 0.0f, lq = 0.0f;
    for (int n = blockIdx.x; n < N; n += gridDim.x) {
        __syncthreads();
        sh[o] = h[(size_t)n * 64 + o];
        __syncthreads();
        int dg = deg[n];
        float acc = cbias[o] + agg[(size_t)n * 64 + o] / (float)max(dg, 1);
        for (int i = 0; i < 64; ++i) acc += sh[i] * root[i * 64 + o];
        mbuf[(size_t)n * 64 + o] = acc;
        ls += acc; lq += acc * acc;
    }
    unsafeAtomicAdd(&bnsum[o], ls);
    unsafeAtomicAdd(&bnsq[o], lq);
}

__global__ void __launch_bounds__(64)
bnfin_kernel(const float* __restrict__ bnsum, const float* __restrict__ bnsq,
             const float* __restrict__ gamma, const float* __restrict__ beta,
             float* __restrict__ scale, float* __restrict__ shift, int N) {
    int o = threadIdx.x;
    float mu  = bnsum[o] / (float)N;
    float var = bnsq[o] / (float)N - mu * mu;
    float inv = rsqrtf(var + 1e-5f);
    float sc  = gamma[o] * inv;
    scale[o] = sc;
    shift[o] = beta[o] - mu * sc;
}

// GRUCell: m_hat = relu(scale*m+shift); h' = (1-z)*n + z*h (in place).
__global__ void __launch_bounds__(64)
gru_kernel(const float* __restrict__ mbuf, const float* __restrict__ scale,
           const float* __restrict__ shift, const float* __restrict__ wih,
           const float* __restrict__ whh, const float* __restrict__ bih,
           const float* __restrict__ bhh, float* __restrict__ h, int N) {
    __shared__ float sm[64], sh[64];
    int d = threadIdx.x;
    for (int n = blockIdx.x; n < N; n += gridDim.x) {
        __syncthreads();
        sm[d] = fmaxf(scale[d] * mbuf[(size_t)n * 64 + d] + shift[d], 0.0f);
        sh[d] = h[(size_t)n * 64 + d];
        __syncthreads();
        float gr = bih[d], gz = bih[64 + d], gn = bih[128 + d];
        float hr = bhh[d], hz = bhh[64 + d], hn = bhh[128 + d];
        for (int j = 0; j < 64; ++j) {
            float mj = sm[j], hj = sh[j];
            gr += wih[(size_t)d         * 64 + j] * mj;
            gz += wih[(size_t)(64 + d)  * 64 + j] * mj;
            gn += wih[(size_t)(128 + d) * 64 + j] * mj;
            hr += whh[(size_t)d         * 64 + j] * hj;
            hz += whh[(size_t)(64 + d)  * 64 + j] * hj;
            hn += whh[(size_t)(128 + d) * 64 + j] * hj;
        }
        float r  = 1.0f / (1.0f + __expf(-(gr + hr)));
        float z  = 1.0f / (1.0f + __expf(-(gz + hz)));
        float nn = tanhf(gn + r * hn);
        h[(size_t)n * 64 + d] = (1.0f - z) * nn + z * sh[d];
    }
}

__global__ void __launch_bounds__(64)
pool_kernel(const float* __restrict__ h, const int* __restrict__ batch,
            float* __restrict__ psum, int* __restrict__ pcnt, int N) {
    int o = threadIdx.x;
    for (int n = blockIdx.x; n < N; n += gridDim.x) {
        int b = batch[n];
        unsafeAtomicAdd(&psum[(size_t)b * 64 + o], h[(size_t)n * 64 + o]);
        if (o == 0) atomicAdd(&pcnt[b], 1);
    }
}

__global__ void __launch_bounds__(64)
head_kernel(const float* __restrict__ psum, const int* __restrict__ pcnt,
            const float* __restrict__ pw, const float* __restrict__ pb,
            const float* __restrict__ ow, const float* __restrict__ ob,
            float* __restrict__ y, int G) {
    __shared__ float sp[64], sr[64];
    int g = blockIdx.x; if (g >= G) return;
    int o = threadIdx.x;
    int c = max(pcnt[g], 1);
    sp[o] = psum[(size_t)g * 64 + o] / (float)c;
    __syncthreads();
    float acc = pb[o];
    for (int i = 0; i < 64; ++i) acc += sp[i] * pw[i * 64 + o];
    sr[o] = fmaxf(acc, 0.0f) * ow[o];
    __syncthreads();
    for (int s = 32; s > 0; s >>= 1) { if (o < s) sr[o] += sr[o + s]; __syncthreads(); }
    if (o == 0) y[g] = sr[0] + ob[0];
}

// ---------------------------------------------------------------------------
static inline size_t align256(size_t x) { return (x + 255) & ~(size_t)255; }

extern "C" void kernel_launch(void* const* d_in, const int* in_sizes, int n_in,
                              void* d_out, int out_size, void* d_ws, size_t ws_size,
                              hipStream_t stream) {
    const float* x      = (const float*)d_in[0];
    const int*   ei     = (const int*)  d_in[1];
    const float* ea     = (const float*)d_in[2];
    const int*   batch  = (const int*)  d_in[3];
    const float* pre_w  = (const float*)d_in[4];
    const float* pre_b  = (const float*)d_in[5];
    const float* ew1    = (const float*)d_in[6];
    const float* eb1    = (const float*)d_in[7];
    const float* ew2    = (const float*)d_in[8];
    const float* eb2    = (const float*)d_in[9];
    const float* root   = (const float*)d_in[10];
    const float* cbias  = (const float*)d_in[11];
    const float* gam    = (const float*)d_in[12];
    const float* bet    = (const float*)d_in[13];
    const float* wih    = (const float*)d_in[14];
    const float* whh    = (const float*)d_in[15];
    const float* bih    = (const float*)d_in[16];
    const float* bhh    = (const float*)d_in[17];
    const float* postw  = (const float*)d_in[18];
    const float* postb  = (const float*)d_in[19];
    const float* outw   = (const float*)d_in[20];
    const float* outb   = (const float*)d_in[21];

    const int N = in_sizes[3];          // 20000
    const int E = in_sizes[1] / 2;      // 100000
    const int G = out_size;             // 128
    const int* src = ei;
    const int* dst = ei + E;

    // workspace carve-up
    char* w = (char*)d_ws;
    size_t off = 0;
    float* hbuf = (float*)(w + off); off += align256((size_t)N * 64 * 4);
    float* mbuf = (float*)(w + off); off += align256((size_t)N * 64 * 4);
    float* agg  = (float*)(w + off); off += align256((size_t)N * 64 * 4);
    unsigned short* h1b = (unsigned short*)(w + off); off += align256((size_t)E * 64 * 2);
    unsigned short* xsb = (unsigned short*)(w + off); off += align256((size_t)E * 64 * 2);
    unsigned* PB = (unsigned*)(w + off); off += align256((size_t)NL * PB_PER_LAYER * 4);
    int*   deg   = (int*)  (w + off); off += align256((size_t)N * 4);
    float* bnsum = (float*)(w + off); off += align256(64 * 4);
    float* bnsq  = (float*)(w + off); off += align256(64 * 4);
    float* bnsc  = (float*)(w + off); off += align256(64 * 4);
    float* bnsh  = (float*)(w + off); off += align256(64 * 4);
    float* psum  = (float*)(w + off); off += align256((size_t)G * 64 * 4);
    int*   pcnt  = (int*)  (w + off); off += align256((size_t)G * 4);
    (void)ws_size; (void)n_in;

    // pre-FC and one-time B packing
    prefc_kernel<<<N, 64, 0, stream>>>(x, pre_w, pre_b, hbuf, N);
    {
        int tot = NL * PB_PER_LAYER;
        pack_w2_kernel<<<(tot + 255) / 256, 256, 0, stream>>>(ew2, eb2, PB);
    }

    const int tiles32 = (E + 31) / 32;
    for (int l = 0; l < NL; ++l) {
        size_t aggN = (size_t)N * 64;
        zero_kernel<<<(int)((aggN + 255) / 256), 256, 0, stream>>>(agg, aggN);
        zero_kernel<<<(N + 255) / 256, 256, 0, stream>>>((float*)deg, (size_t)N);
        zero_kernel<<<1, 256, 0, stream>>>(bnsum, 64);
        zero_kernel<<<1, 256, 0, stream>>>(bnsq, 64);

        edge_pre_kernel<<<E, 64, 0, stream>>>(ea, ew1 + (size_t)l * FEE * 64,
                                              eb1 + (size_t)l * 64, src, hbuf,
                                              h1b, xsb, E);
        deg_kernel<<<(E + 255) / 256, 256, 0, stream>>>(dst, deg, E);
        msg_wmma_kernel<<<tiles32, 32, 0, stream>>>(h1b, xsb,
                                                    PB + (size_t)l * PB_PER_LAYER,
                                                    dst, agg, E);
        node_m_kernel<<<512, 64, 0, stream>>>(agg, deg, hbuf,
                                              root + (size_t)l * 64 * 64,
                                              cbias + (size_t)l * 64,
                                              mbuf, bnsum, bnsq, N);
        bnfin_kernel<<<1, 64, 0, stream>>>(bnsum, bnsq, gam + (size_t)l * 64,
                                           bet + (size_t)l * 64, bnsc, bnsh, N);
        gru_kernel<<<512, 64, 0, stream>>>(mbuf, bnsc, bnsh,
                                           wih + (size_t)l * 192 * 64,
                                           whh + (size_t)l * 192 * 64,
                                           bih + (size_t)l * 192,
                                           bhh + (size_t)l * 192, hbuf, N);
    }

    zero_kernel<<<(int)(((size_t)G * 64 + 255) / 256), 256, 0, stream>>>(psum, (size_t)G * 64);
    zero_kernel<<<1, 256, 0, stream>>>((float*)pcnt, (size_t)G);
    pool_kernel<<<512, 64, 0, stream>>>(hbuf, batch, psum, pcnt, N);
    head_kernel<<<G, 64, 0, stream>>>(psum, pcnt, postw, postb, outw, outb,
                                      (float*)d_out, G);
}
